// SuperFormer_18468359373037
// MI455X (gfx1250) — matrix operand order repaired and checked by reading, not compiled, not run on hardware
//
#include <hip/hip_runtime.h>

// ---------------------------------------------------------------------------
// Types / helpers
// ---------------------------------------------------------------------------
typedef unsigned short u16;
typedef __bf16 bf16x16 __attribute__((ext_vector_type(16)));
typedef float  f32x8   __attribute__((ext_vector_type(8)));
typedef unsigned int u32x4 __attribute__((ext_vector_type(4)));

union Frag { bf16x16 v; u32x4 q[2]; };

__device__ __forceinline__ float bf2f(u16 h) {
    return __uint_as_float(((unsigned int)h) << 16);
}
__device__ __forceinline__ u16 f2bf(float f) {
    unsigned int u = __float_as_uint(f);
    u += 0x7fffu + ((u >> 16) & 1u);
    return (u16)(u >> 16);
}

#define TM 64
#define TN 64
#define APAD 40   // padded A-row stride in u16 (80B: kills LDS bank conflicts,
                  // keeps every fragment segment 16B-aligned)

// ---------------------------------------------------------------------------
// Weight convert + reorder: f32 [Cout][Cin][3][3] -> bf16 [Cout][tap][Cin]
// so GEMM-K = tap*Cin + ci matches NHWC activation chunks.
// ---------------------------------------------------------------------------
__global__ __launch_bounds__(256)
void sf_cvt_w(const float* __restrict__ src, u16* __restrict__ dst,
              int Cout, int Cin) {
    int i = blockIdx.x * 256 + threadIdx.x;
    int total = Cout * Cin * 9;
    if (i >= total) return;
    int co = i / (Cin * 9);
    int r  = i - co * Cin * 9;
    int ci = r / 9;
    int tap = r - ci * 9;
    dst[(size_t)co * Cin * 9 + tap * Cin + ci] = f2bf(src[i]);
}

// ---------------------------------------------------------------------------
// Direct 3x3 SAME conv for tiny Cin (1/2): NCHW f32 in -> NHWC bf16 out
// (+bias), accumulates per-channel sum/sumsq.  grid:(ceil(HW/256),Cout,B)
// ---------------------------------------------------------------------------
template <int Cin, int HH>
__global__ __launch_bounds__(256)
void sf_conv_small(const float* __restrict__ X, const float* __restrict__ Wgt,
                   const float* __restrict__ bias, u16* __restrict__ Y,
                   float* __restrict__ gsum, float* __restrict__ gsumsq,
                   int Cout) {
    constexpr int H = HH, W = HH, HW = H * W;
    const int n = blockIdx.z, co = blockIdx.y;
    const int p = blockIdx.x * 256 + threadIdx.x;
    float val = 0.f;
    bool ok = p < HW;
    if (ok) {
        int h = p / W, w = p - h * W;
        float a = bias[co];
        #pragma unroll
        for (int ci = 0; ci < Cin; ++ci)
            #pragma unroll
            for (int dh = -1; dh <= 1; ++dh)
                #pragma unroll
                for (int dw = -1; dw <= 1; ++dw) {
                    int hh = h + dh, ww = w + dw;
                    if (hh >= 0 && hh < H && ww >= 0 && ww < W)
                        a += X[((size_t)n * Cin + ci) * HW + hh * W + ww] *
                             Wgt[((co * Cin + ci) * 3 + (dh + 1)) * 3 + (dw + 1)];
                }
        val = a;
        Y[((size_t)n * HW + p) * Cout + co] = f2bf(val);   // NHWC
    }
    __shared__ float red[2];
    if (threadIdx.x == 0) { red[0] = 0.f; red[1] = 0.f; }
    __syncthreads();
    atomicAdd(&red[0], ok ? val : 0.f);
    atomicAdd(&red[1], ok ? val * val : 0.f);
    __syncthreads();
    if (threadIdx.x == 0) {
        atomicAdd(&gsum[co], red[0]);
        atomicAdd(&gsumsq[co], red[1]);
    }
}

// ---------------------------------------------------------------------------
// Implicit-GEMM 3x3 SAME conv, NHWC bf16 -> NHWC bf16, WMMA bf16->f32.
//   M = B*H*W spatial, N = Cout, K = 9*Cin ordered tap-major (tap*Cin+ci).
//   Block 256 thr = 8 waves (4M x 2N); wave: 16 rows x 2x16 couts (2 acc).
//   A tile (64x32 bf16, padded rows) double-buffered in LDS, filled with
//   global_load_async_to_lds_b128; next stage overlaps current WMMAs.
//   All shape params compile-time -> divisions become shifts/mul-shifts.
// grid: (S/TM, Cout/TN)
// ---------------------------------------------------------------------------
template <int Cin, int Cout, int HH>
__global__ __launch_bounds__(256)
void sf_conv_wmma(const u16* __restrict__ X, const u16* __restrict__ Wgt,
                  const float* __restrict__ bias, u16* __restrict__ Y,
                  float* __restrict__ gsum, float* __restrict__ gsumsq) {
    constexpr int B = 32, H = HH, W = HH;
    constexpr int HW = H * W;
    constexpr int S  = B * HW;
    constexpr int K  = Cin * 9;
    constexpr int nsteps = K >> 5;          // even for Cin in {64,128,256}

    const int s0 = blockIdx.x * TM;
    const int n0 = blockIdx.y * TN;

    const int t    = threadIdx.x;
    const int lane = t & 31;
    const int wid  = t >> 5;
    const int mw   = wid & 3;   // 4 M-waves
    const int nw   = wid >> 2;  // 2 N-waves
    const int lh   = lane >> 4;
    const int lm   = lane & 15;

    __shared__ __align__(16) u16 Alds[2][TM][APAD];
    __shared__ float cs[TN], css[TN];
    if (t < TN) { cs[t] = 0.f; css[t] = 0.f; }

    // ---- staging role: thread t copies one 16B segment of one A row ----
    const int m  = t >> 2;        // row 0..63
    const int sg = (t & 3) * 8;   // channel sub-offset (8 bf16 = 16B)
    const int sA = s0 + m;
    const bool mvalid = sA < S;
    const int nA = sA / HW;                   // const divisors -> mul-shift
    const int hw = sA - nA * HW;
    const int hA = hw / W;
    const int wA = hw - hA * W;
    const u16* xrow = X + (size_t)nA * HW * Cin;

    auto stage = [&](int k0, int buf) {
        int tap = k0 / Cin;                   // Cin is a power of two: shift
        int ci0 = k0 - tap * Cin;
        int dh = tap / 3 - 1, dw = tap - (tap / 3) * 3 - 1;
        int hh = hA + dh, ww = wA + dw;
        u16* dst = &Alds[buf][m][sg];
        if (mvalid && hh >= 0 && hh < H && ww >= 0 && ww < W) {
            const u16* src = xrow + ((size_t)hh * W + ww) * Cin + ci0 + sg;
            unsigned lo = (unsigned)(size_t)(void*)dst;  // low 32 = LDS offset
            asm volatile("global_load_async_to_lds_b128 %0, %1, off"
                         :: "v"(lo), "v"(src) : "memory");
        } else {
            u32x4 z = {0u, 0u, 0u, 0u};
            *(u32x4*)dst = z;                 // zero-pad border rows
        }
    };

    const int co0 = n0 + nw * 32 + lm;
    const int co1 = co0 + 16;
    const u16* w0 = Wgt + (size_t)co0 * K + (lh ? 8 : 0);
    const u16* w1 = Wgt + (size_t)co1 * K + (lh ? 8 : 0);

    f32x8 acc0 = {0.f, 0.f, 0.f, 0.f, 0.f, 0.f, 0.f, 0.f};
    f32x8 acc1 = {0.f, 0.f, 0.f, 0.f, 0.f, 0.f, 0.f, 0.f};

    stage(0, 0);
    #pragma unroll 2
    for (int i = 0; i < nsteps; ++i) {
        asm volatile("s_wait_asynccnt 0x0" ::: "memory");
        __syncthreads();                       // A[i&1] visible to all waves
        if (i + 1 < nsteps) stage((i + 1) << 5, (i + 1) & 1);  // overlap

        const int k0 = i << 5;
        Frag a, b0, b1;
        const u16* arow = &Alds[i & 1][mw * 16 + lm][0];
        a.q[0]  = *(const u32x4*)(arow + (lh ? 8 : 0));
        a.q[1]  = *(const u32x4*)(arow + (lh ? 24 : 16));
        b0.q[0] = *(const u32x4*)(w0 + k0);
        b0.q[1] = *(const u32x4*)(w0 + k0 + 16);
        b1.q[0] = *(const u32x4*)(w1 + k0);
        b1.q[1] = *(const u32x4*)(w1 + k0 + 16);
        __builtin_prefetch(w0 + k0 + 32, 0, 1);
        __builtin_prefetch(w1 + k0 + 32, 0, 1);

        acc0 = __builtin_amdgcn_wmma_f32_16x16x32_bf16(
            false, a.v, false, b0.v, (short)0, acc0, false, false);
        acc1 = __builtin_amdgcn_wmma_f32_16x16x32_bf16(
            false, a.v, false, b1.v, (short)0, acc1, false, false);
    }

    // ---- epilogue: bias, NHWC bf16 store, per-channel stats --------------
    const float b0v = bias[co0];
    const float b1v = bias[co1];
    float sa0 = 0.f, sq0 = 0.f, sa1 = 0.f, sq1 = 0.f;
    #pragma unroll
    for (int v = 0; v < 8; ++v) {
        int mm = mw * 16 + v + lh * 8;     // C/D layout: VGPR v, lane-half
        int s  = s0 + mm;
        if (s < S) {
            float x0 = acc0[v] + b0v;
            float x1 = acc1[v] + b1v;
            Y[(size_t)s * Cout + co0] = f2bf(x0);
            Y[(size_t)s * Cout + co1] = f2bf(x1);
            sa0 += x0; sq0 += x0 * x0;
            sa1 += x1; sq1 += x1 * x1;
        }
    }
    atomicAdd(&cs[nw * 32 + lm], sa0);
    atomicAdd(&css[nw * 32 + lm], sq0);
    atomicAdd(&cs[nw * 32 + 16 + lm], sa1);
    atomicAdd(&css[nw * 32 + 16 + lm], sq1);
    __syncthreads();
    if (t < TN) {
        atomicAdd(&gsum[n0 + t], cs[t]);
        atomicAdd(&gsumsq[n0 + t], css[t]);
    }
}

// ---------------------------------------------------------------------------
// BN (+ReLU) and optional 2x2 maxpool, NHWC bf16 -> NHWC bf16
// ---------------------------------------------------------------------------
__global__ __launch_bounds__(256)
void sf_bn_act(const u16* __restrict__ Y, const float* __restrict__ gsum,
               const float* __restrict__ gsumsq, const float* __restrict__ g,
               const float* __restrict__ be, u16* __restrict__ O,
               int B, int C, int H, int W, int pool, float invcnt) {
    const int Ho = pool ? (H >> 1) : H;
    const int Wo = pool ? (W >> 1) : W;
    size_t total = (size_t)B * C * Ho * Wo;
    size_t i = (size_t)blockIdx.x * 256 + threadIdx.x;
    if (i >= total) return;
    int c  = (int)(i % C); size_t r = i / C;
    int wo = (int)(r % Wo); r /= Wo;
    int ho = (int)(r % Ho);
    int n  = (int)(r / Ho);
    float mn  = gsum[c] * invcnt;
    float var = gsumsq[c] * invcnt - mn * mn;
    float sc  = g[c] * rsqrtf(var + 1e-5f);
    float sh  = be[c] - mn * sc;
    float o;
    if (pool) {
        int h = ho * 2, w = wo * 2;
        const u16* base = Y + (((size_t)n * H + h) * W + w) * C + c;
        float v0 = bf2f(base[0])                   * sc + sh;
        float v1 = bf2f(base[C])                   * sc + sh;
        float v2 = bf2f(base[(size_t)W * C])       * sc + sh;
        float v3 = bf2f(base[(size_t)(W + 1) * C]) * sc + sh;
        v0 = v0 > 0.f ? v0 : 0.f;  v1 = v1 > 0.f ? v1 : 0.f;
        v2 = v2 > 0.f ? v2 : 0.f;  v3 = v3 > 0.f ? v3 : 0.f;
        o = fmaxf(fmaxf(v0, v1), fmaxf(v2, v3));
    } else {
        float x = bf2f(Y[(((size_t)n * H + ho) * W + wo) * C + c]) * sc + sh;
        o = x > 0.f ? x : 0.f;
    }
    O[i] = f2bf(o);
}

// ---------------------------------------------------------------------------
// BN + ReLU + global average pool (NHWC, C==256) -> f32 [B,C]. grid=B
// Threads map to channels -> contiguous (coalesced) reads per iteration.
// ---------------------------------------------------------------------------
__global__ __launch_bounds__(256)
void sf_bn_avgpool(const u16* __restrict__ Y, const float* __restrict__ gsum,
                   const float* __restrict__ gsumsq, const float* __restrict__ g,
                   const float* __restrict__ be, float* __restrict__ feat,
                   int C, int H, int W, float invcnt) {
    const int n = blockIdx.x, c = threadIdx.x;
    float mn  = gsum[c] * invcnt;
    float var = gsumsq[c] * invcnt - mn * mn;
    float sc  = g[c] * rsqrtf(var + 1e-5f);
    float sh  = be[c] - mn * sc;
    const u16* base = Y + (size_t)n * H * W * C + c;
    float acc = 0.f;
    for (int p = 0; p < H * W; ++p) {
        float x = bf2f(base[(size_t)p * C]) * sc + sh;
        acc += x > 0.f ? x : 0.f;
    }
    feat[n * C + c] = acc / (float)(H * W);
}

// ---------------------------------------------------------------------------
// Per-sample 2-layer GCN (25 nodes, 48 edges + self loops), BN over nodes,
// ReLU, max-over-nodes. grid = B, block = 256.
// ---------------------------------------------------------------------------
__global__ __launch_bounds__(256)
void sf_gcn(const float* __restrict__ X, const int* __restrict__ EI,
            const float* __restrict__ W1, const float* __restrict__ b1,
            const float* __restrict__ g1, const float* __restrict__ be1,
            const float* __restrict__ W2, const float* __restrict__ b2,
            const float* __restrict__ g2, const float* __restrict__ be2,
            float* __restrict__ feat) {
    const int N = 25, E = 48, NE = 73;
    const int bs = blockIdx.x, t = threadIdx.x;
    __shared__ float xs[25 * 3];
    __shared__ float deg[25];
    __shared__ float nrm[73];
    __shared__ float t1[25 * 64];
    __shared__ float a1[25 * 64];
    __shared__ float a2[25 * 256];

    if (t < 75) xs[t] = X[bs * 75 + t];
    if (t < 25) deg[t] = 0.f;
    __syncthreads();
    if (t < NE) {
        int d = (t < E) ? EI[E + t] : (t - E);
        atomicAdd(&deg[d], 1.f);
    }
    __syncthreads();
    if (t < NE) {
        int s = (t < E) ? EI[t] : (t - E);
        int d = (t < E) ? EI[E + t] : (t - E);
        nrm[t] = rsqrtf(deg[s]) * rsqrtf(deg[d]);
    }
    for (int idx = t; idx < N * 64; idx += 256) {
        int node = idx >> 6, f = idx & 63;
        float acc = b1[f];
        #pragma unroll
        for (int j = 0; j < 3; ++j) acc += xs[node * 3 + j] * W1[j * 64 + f];
        t1[idx] = acc;
        a1[idx] = 0.f;
    }
    __syncthreads();
    if (t < 64) {
        for (int e = 0; e < NE; ++e) {
            int s = (e < E) ? EI[e] : (e - E);
            int d = (e < E) ? EI[E + e] : (e - E);
            a1[d * 64 + t] += t1[s * 64 + t] * nrm[e];
        }
    }
    __syncthreads();
    if (t < 64) {
        float m = 0.f, v = 0.f;
        for (int n2 = 0; n2 < N; ++n2) m += a1[n2 * 64 + t];
        m /= N;
        for (int n2 = 0; n2 < N; ++n2) { float x = a1[n2 * 64 + t] - m; v += x * x; }
        v /= N;
        float sc = g1[t] * rsqrtf(v + 1e-5f), sh = be1[t] - m * sc;
        for (int n2 = 0; n2 < N; ++n2) {
            float x = a1[n2 * 64 + t] * sc + sh;
            a1[n2 * 64 + t] = x > 0.f ? x : 0.f;
        }
    }
    for (int idx = t; idx < N * 256; idx += 256) a2[idx] = 0.f;
    __syncthreads();
    {
        for (int e = 0; e < NE; ++e) {
            int s = (e < E) ? EI[e] : (e - E);
            int d = (e < E) ? EI[E + e] : (e - E);
            float h = b2[t];
            for (int j = 0; j < 64; ++j) h += a1[s * 64 + j] * W2[j * 256 + t];
            a2[d * 256 + t] += h * nrm[e];
        }
    }
    __syncthreads();
    {
        float m = 0.f, v = 0.f;
        for (int n2 = 0; n2 < N; ++n2) m += a2[n2 * 256 + t];
        m /= N;
        for (int n2 = 0; n2 < N; ++n2) { float x = a2[n2 * 256 + t] - m; v += x * x; }
        v /= N;
        float sc = g2[t] * rsqrtf(v + 1e-5f), sh = be2[t] - m * sc;
        float best = -3.4e38f;
        for (int n2 = 0; n2 < N; ++n2) {
            float x = a2[n2 * 256 + t] * sc + sh;
            best = x > best ? x : best;
        }
        feat[bs * 256 + t] = best;
    }
}

// ---------------------------------------------------------------------------
// combined = cat(mf,ff,sf); out = (c0*mf + c1*ff + c2*sf) @ Wfc + bfc
// ---------------------------------------------------------------------------
__global__ __launch_bounds__(256)
void sf_combine_fc(const float* __restrict__ mf, const float* __restrict__ ff,
                   const float* __restrict__ sf, const float* __restrict__ Wfc,
                   const float* __restrict__ bfc, float* __restrict__ out, int NC) {
    const int n = blockIdx.x, t = threadIdx.x;
    __shared__ float r[256];
    float a0 = mf[n * 256 + 0], a1 = mf[n * 256 + 1], a2 = mf[n * 256 + 2];
    r[t] = a0 * mf[n * 256 + t] + a1 * ff[n * 256 + t] + a2 * sf[n * 256 + t];
    __syncthreads();
    for (int co = t; co < NC; co += 256) {
        float acc = bfc[co];
        for (int j = 0; j < 256; ++j) acc += r[j] * Wfc[j * NC + co];
        out[n * NC + co] = acc;
    }
}

// ---------------------------------------------------------------------------
// Host orchestration
// ---------------------------------------------------------------------------
extern "C" void kernel_launch(void* const* d_in, const int* in_sizes, int n_in,
                              void* d_out, int out_size, void* d_ws, size_t ws_size,
                              hipStream_t stream) {
    const float* x_mask = (const float*)d_in[0];   // [32,1,224,224]
    const float* x_flow = (const float*)d_in[1];   // [32,2,224,224]
    const float* x_skel = (const float*)d_in[2];   // [32,25,3]
    const int*   edge   = (const int*)d_in[3];     // [2,48]
    const float* const* P = (const float* const*)(d_in + 4);

    const int B = 32, NC = 400;

    char* ws = (char*)d_ws;
    size_t off = 0;
    auto carve = [&](size_t bytes) -> void* {
        void* p = ws + off;
        off += (bytes + 255) & ~(size_t)255;
        return p;
    };
    u16* Wb_m2 = (u16*)carve((size_t)73728 * 2);
    u16* Wb_m3 = (u16*)carve((size_t)294912 * 2);
    u16* Wb_f2 = (u16*)carve((size_t)73728 * 2);
    u16* Wb_f3 = (u16*)carve((size_t)294912 * 2);
    u16* Wb_f4 = (u16*)carve((size_t)589824 * 2);
    float* stats = (float*)carve(512 * 4);
    float* featM = (float*)carve((size_t)B * 256 * 4);
    float* featF = (float*)carve((size_t)B * 256 * 4);
    float* featS = (float*)carve((size_t)B * 256 * 4);
    u16* bufA = (u16*)carve((size_t)102760448 * 2);
    u16* bufB = (u16*)carve((size_t)25690112 * 2);
    float* gsum = stats, *gss = stats + 256;

    auto cvtw = [&](const float* s, u16* d, int Cout, int Cin) {
        int n = Cout * Cin * 9;
        sf_cvt_w<<<(n + 255) / 256, 256, 0, stream>>>(s, d, Cout, Cin);
    };
    cvtw(P[4],  Wb_m2, 128, 64);
    cvtw(P[8],  Wb_m3, 256, 128);
    cvtw(P[16], Wb_f2, 128, 64);
    cvtw(P[20], Wb_f3, 256, 128);
    cvtw(P[24], Wb_f4, 256, 256);

    auto zstats = [&]() { hipMemsetAsync(stats, 0, 512 * 4, stream); };
    auto bn_act = [&](const u16* Y, const float* g, const float* be, u16* O,
                      int C, int H, int W, int pool) {
        int Ho = pool ? H / 2 : H, Wo = pool ? W / 2 : W;
        size_t total = (size_t)B * C * Ho * Wo;
        sf_bn_act<<<(unsigned)((total + 255) / 256), 256, 0, stream>>>(
            Y, gsum, gss, g, be, O, B, C, H, W, pool, 1.f / ((float)B * H * W));
    };
    auto bn_avg = [&](const u16* Y, const float* g, const float* be, float* feat,
                      int C, int H, int W) {
        sf_bn_avgpool<<<B, 256, 0, stream>>>(Y, gsum, gss, g, be, feat,
                                             C, H, W, 1.f / ((float)B * H * W));
    };

    dim3 gSmall((224 * 224 + 255) / 256, 64, B);
    dim3 gC1(32 * 112 * 112 / TM, 128 / TN);   // 64->128 @112
    dim3 gC2(32 * 56 * 56 / TM, 256 / TN);     // 128->256 @56 and 256->256 @56

    // ================= mask branch =================
    zstats();
    sf_conv_small<1, 224><<<gSmall, 256, 0, stream>>>(x_mask, P[0], P[1], bufA,
                                                      gsum, gss, 64);
    bn_act(bufA, P[2], P[3], bufB, 64, 224, 224, 1);        // -> [32,112,112,64]
    zstats();
    sf_conv_wmma<64, 128, 112><<<gC1, 256, 0, stream>>>(bufB, Wb_m2, P[5], bufA,
                                                        gsum, gss);
    bn_act(bufA, P[6], P[7], bufB, 128, 112, 112, 1);       // -> [32,56,56,128]
    zstats();
    sf_conv_wmma<128, 256, 56><<<gC2, 256, 0, stream>>>(bufB, Wb_m3, P[9], bufA,
                                                        gsum, gss);
    bn_avg(bufA, P[10], P[11], featM, 256, 56, 56);         // -> [32,256]

    // ================= flow branch =================
    zstats();
    sf_conv_small<2, 224><<<gSmall, 256, 0, stream>>>(x_flow, P[12], P[13], bufA,
                                                      gsum, gss, 64);
    bn_act(bufA, P[14], P[15], bufB, 64, 224, 224, 1);
    zstats();
    sf_conv_wmma<64, 128, 112><<<gC1, 256, 0, stream>>>(bufB, Wb_f2, P[17], bufA,
                                                        gsum, gss);
    bn_act(bufA, P[18], P[19], bufB, 128, 112, 112, 1);
    zstats();
    sf_conv_wmma<128, 256, 56><<<gC2, 256, 0, stream>>>(bufB, Wb_f3, P[21], bufA,
                                                        gsum, gss);
    bn_act(bufA, P[22], P[23], bufB, 256, 56, 56, 0);       // full-res relu(bn)
    zstats();
    sf_conv_wmma<256, 256, 56><<<gC2, 256, 0, stream>>>(bufB, Wb_f4, P[25], bufA,
                                                        gsum, gss);
    bn_avg(bufA, P[26], P[27], featF, 256, 56, 56);

    // ================= GCN branch =================
    sf_gcn<<<B, 256, 0, stream>>>(x_skel, edge, P[28], P[29], P[30], P[31],
                                  P[32], P[33], P[34], P[35], featS);

    // ================= combine + FC =================
    sf_combine_fc<<<B, 256, 0, stream>>>(featM, featF, featS, P[36], P[37],
                                         (float*)d_out, NC);
}
